// Block_824633721573
// MI455X (gfx1250) — compile-verified, matrix operand order checked by reading, not run on hardware
//
#include <hip/hip_runtime.h>
#include <hip/hip_bf16.h>

#define C_DIM 192
#define HD 32
#define NH 6
#define WS 8
#define SHIFT 4

typedef __attribute__((ext_vector_type(16))) __bf16 v16bf;
typedef __attribute__((ext_vector_type(8)))  float  v8f;

// Native hardware f32->bf16 conversion (RNE on gfx1250)
__device__ __forceinline__ unsigned short f2bf(float f) {
    __bf16 b = (__bf16)f;
    return __builtin_bit_cast(unsigned short, b);
}
__device__ __forceinline__ unsigned pack2(float a, float b) {
    return (unsigned)f2bf(a) | ((unsigned)f2bf(b) << 16);
}
__device__ __forceinline__ __bf16 bfbits(unsigned short s) {
    return __builtin_bit_cast(__bf16, s);
}

__device__ __forceinline__ v8f wmma_bf16(v16bf a, v16bf b, v8f c) {
    return __builtin_amdgcn_wmma_f32_16x16x32_bf16(
        /*neg_a=*/false, a, /*neg_b=*/false, b,
        /*c_mod=*/(short)0, c, /*reuse_a=*/false, /*reuse_b=*/false);
}

// A or B fragment: lane holds row/col (base + lane&15), K contiguous in memory.
// element(lane, j) = W[(rc + lane%16)*ld + kBase + Kmap(lane, j)]
// Kmap: lanes 0-15 -> K = j (j<8), 16+(j-8) (j>=8); lanes 16-31 -> +8.
__device__ __forceinline__ v16bf load_frag_rowK(const unsigned short* W, int ld,
                                                int rcBase, int kBase) {
    const int lane = threadIdx.x & 31;
    const unsigned short* p = W + (rcBase + (lane & 15)) * ld + kBase + ((lane >> 4) << 3);
    v16bf f;
#pragma unroll
    for (int j = 0; j < 8; ++j) f[j] = bfbits(p[j]);
#pragma unroll
    for (int j = 0; j < 8; ++j) f[8 + j] = bfbits(p[16 + j]);
    return f;
}

// ---------------------------------------------------------------------------
// Kernel 0: fp32 -> bf16 weight conversion into workspace
// ---------------------------------------------------------------------------
#define N_QKV (576 * 192)
#define N_LIN (192 * 192)
#define N_W1  (768 * 192)
#define N_W2  (192 * 768)

__global__ void prep_weights(const float* __restrict__ qkv_w,
                             const float* __restrict__ lin_w,
                             const float* __restrict__ w1,
                             const float* __restrict__ w2,
                             unsigned short* __restrict__ out) {
    int i = blockIdx.x * blockDim.x + threadIdx.x;
    if (i < N_QKV)                          out[i] = f2bf(qkv_w[i]);
    else if (i < N_QKV + N_LIN)             out[i] = f2bf(lin_w[i - N_QKV]);
    else if (i < N_QKV + N_LIN + N_W1)      out[i] = f2bf(w1[i - N_QKV - N_LIN]);
    else if (i < N_QKV + N_LIN + N_W1 + N_W2)
                                            out[i] = f2bf(w2[i - N_QKV - N_LIN - N_W1]);
}

// ---------------------------------------------------------------------------
// Kernel 1: LN1 + shifted-window attention + output proj + residual -> x1
// One workgroup (256 thr = 8 wave32) per (batch, window).
// Padded LDS layout (bytes), heads processed 2 per pass (3 passes):
//   sc   f32 [128][68]  @ 0       34816   scores (2 heads)
//   pb   u16 [128][72]  @ 34816   18432   softmax probs bf16
//   qkvb u16 [64][392]  @ 53248   50176   q (0..191), k (192..383) per token
//   vt   u16 [192][72]  @ 103424  27648   V transposed: [chan][token]
//   yb   u16 [64][200]  @ 131072  25600   LN1 out, reused as attn out
// total 156672 B
// ---------------------------------------------------------------------------
#define LD_Y   200
#define LD_QK  392
#define LD_VT  72
#define LD_SC  68
#define LD_PB  72
#define LD_H   776

__global__ __launch_bounds__(256, 1)
void swin_attn(const float* __restrict__ x,
               const float* __restrict__ ln1_g, const float* __restrict__ ln1_b,
               const unsigned short* __restrict__ wqkv,
               const float* __restrict__ qkv_b,
               const float* __restrict__ rpp,
               const unsigned short* __restrict__ wlin,
               const float* __restrict__ lin_b,
               float* __restrict__ x1) {
    extern __shared__ __align__(16) char smem[];
    float*          sc   = (float*)smem;
    unsigned short* pb   = (unsigned short*)(smem + 34816);
    unsigned short* qkvb = (unsigned short*)(smem + 53248);
    unsigned short* vt   = (unsigned short*)(smem + 103424);
    unsigned short* yb   = (unsigned short*)(smem + 131072);

    const int tid  = threadIdx.x;
    const int lane = tid & 31;
    // wave id as an SGPR so tile loops use scalar branches (not exec masking)
    const int wv   = __builtin_amdgcn_readfirstlane(tid >> 5);
    const int blk  = blockIdx.x;
    const int bat  = blk >> 6;
    const int w    = blk & 63;
    const int wh   = w >> 3;
    const int wc   = w & 7;

    if (tid == 0) { __builtin_prefetch(wqkv, 0, 0); __builtin_prefetch(wlin, 0, 0); }

    // ---- Phase 1: shifted gather + LN1 (4 lanes per token, shuffle-reduced) ----
    {
        const int tok = tid >> 2, part = tid & 3;       // 64 tokens x 4 parts(48ch)
        const int i1 = tok >> 3, j1 = tok & 7;
        const int hs  = ((wh << 3) + i1 + SHIFT) & 63;
        const int ws_ = ((wc << 3) + j1 + SHIFT) & 63;
        const float* src = x + ((long)((bat << 6) + hs) * 64 + ws_) * C_DIM;
        const float4* s4 = (const float4*)(src + part * 48);
        float4 vals[12];
        float s = 0.f, s2 = 0.f;
#pragma unroll
        for (int j = 0; j < 12; ++j) {
            float4 v = s4[j]; vals[j] = v;
            s  += v.x + v.y + v.z + v.w;
            s2 += v.x * v.x + v.y * v.y + v.z * v.z + v.w * v.w;
        }
        s  += __shfl_xor(s, 1);  s  += __shfl_xor(s, 2);
        s2 += __shfl_xor(s2, 1); s2 += __shfl_xor(s2, 2);
        const float m = s * (1.f / C_DIM);
        const float r = rsqrtf(s2 * (1.f / C_DIM) - m * m + 1e-5f);
        unsigned* dst = (unsigned*)(yb + tok * LD_Y + part * 48);
        const float4* g4 = (const float4*)(ln1_g + part * 48);
        const float4* b4 = (const float4*)(ln1_b + part * 48);
#pragma unroll
        for (int j = 0; j < 12; ++j) {
            float4 v = vals[j], g = g4[j], bb = b4[j];
            dst[2 * j]     = pack2((v.x - m) * r * g.x + bb.x, (v.y - m) * r * g.y + bb.y);
            dst[2 * j + 1] = pack2((v.z - m) * r * g.z + bb.z, (v.w - m) * r * g.w + bb.w);
        }
    }
    __syncthreads();

    // ---- Phase 2: QKV GEMM (64x192)@(192x576); V stored transposed ----
    {
        const int mT = wv & 3;
        v16bf afr[6];
#pragma unroll
        for (int kc = 0; kc < 6; ++kc)
            afr[kc] = load_frag_rowK(yb, LD_Y, mT << 4, kc << 5);
        for (int nT = (wv >> 2); nT < 36; nT += 2) {
            v16bf bfr[6];
#pragma unroll
            for (int kc = 0; kc < 6; ++kc)
                bfr[kc] = load_frag_rowK(wqkv, C_DIM, nT << 4, kc << 5);
            v8f acc = {};
#pragma unroll
            for (int kc = 0; kc < 6; ++kc) acc = wmma_bf16(afr[kc], bfr[kc], acc);
            const int col = (nT << 4) + (lane & 15);
            const float bias = qkv_b[col];
            const int mo = (lane >> 4) << 3;
            if (nT < 24) {                      // q,k -> qkvb[token][col]
#pragma unroll
                for (int r2 = 0; r2 < 8; ++r2)
                    qkvb[((mT << 4) + mo + r2) * LD_QK + col] = f2bf(acc[r2] + bias);
            } else {                            // v -> vt[chan][token], 16B store
                uint4 pk;
                pk.x = pack2(acc[0] + bias, acc[1] + bias);
                pk.y = pack2(acc[2] + bias, acc[3] + bias);
                pk.z = pack2(acc[4] + bias, acc[5] + bias);
                pk.w = pack2(acc[6] + bias, acc[7] + bias);
                *(uint4*)(vt + (col - 384) * LD_VT + (mT << 4) + mo) = pk;
            }
        }
    }
    __syncthreads();

    const float scale = 0.17677669529663687f;  // 1/sqrt(32)

    // ---- Phases 3-5: attention, 2 heads per pass (3 passes) ----
    for (int hp = 0; hp < 3; ++hp) {
        // scores: S = q @ k^T * scale + rel_bias, shift mask
        {
            const int hl = wv >> 2, head = hp * 2 + hl, mT = wv & 3;
            v16bf a = load_frag_rowK(qkvb, LD_QK, mT << 4, head << 5);
            for (int nT = 0; nT < 4; ++nT) {
                v16bf bb = load_frag_rowK(qkvb, LD_QK, nT << 4, C_DIM + (head << 5));
                v8f acc = {};
                acc = wmma_bf16(a, bb, acc);
                const int kq = (nT << 4) + (lane & 15);
                const int ki = kq >> 3, kj = kq & 7;
                const int mo = (lane >> 4) << 3;
#pragma unroll
                for (int r2 = 0; r2 < 8; ++r2) {
                    const int p  = (mT << 4) + mo + r2;
                    const int pi = p >> 3, pj = p & 7;
                    float v = acc[r2] * scale
                            + rpp[head * 225 + (pi - ki + 7) * 15 + (pj - kj + 7)];
                    const bool msk = ((wh == 7) && ((pi < 4) != (ki < 4))) ||
                                     ((wc == 7) && ((pj < 4) != (kj < 4)));
                    sc[((hl << 6) + p) * LD_SC + kq] = msk ? -1e30f : v;
                }
            }
        }
        __syncthreads();

        // softmax (128 rows) -> bf16 probs in pb
        if (tid < 128) {
            float4* row4 = (float4*)(sc + tid * LD_SC);
            float mx = -1e30f;
#pragma unroll
            for (int j = 0; j < 16; ++j) {
                float4 v = row4[j];
                mx = fmaxf(mx, fmaxf(fmaxf(v.x, v.y), fmaxf(v.z, v.w)));
            }
            float s = 0.f;
#pragma unroll
            for (int j = 0; j < 16; ++j) {
                float4 v = row4[j];
                v.x = __expf(v.x - mx); v.y = __expf(v.y - mx);
                v.z = __expf(v.z - mx); v.w = __expf(v.w - mx);
                s += v.x + v.y + v.z + v.w;
                row4[j] = v;
            }
            const float inv = 1.f / s;
            unsigned* dst = (unsigned*)(pb + tid * LD_PB);
#pragma unroll
            for (int j = 0; j < 16; ++j) {
                float4 v = row4[j];
                dst[2 * j]     = pack2(v.x * inv, v.y * inv);
                dst[2 * j + 1] = pack2(v.z * inv, v.w * inv);
            }
        }
        __syncthreads();

        // out = P @ V  (64x64 @ 64x32 per head) -> yb (attn out, bf16)
        {
            const int hl = wv >> 2, head = hp * 2 + hl, mT = wv & 3;
            v16bf pa[2];
#pragma unroll
            for (int kc = 0; kc < 2; ++kc)
                pa[kc] = load_frag_rowK(pb, LD_PB, (hl << 6) + (mT << 4), kc << 5);
            for (int nT = 0; nT < 2; ++nT) {
                v16bf bf0 = load_frag_rowK(vt, LD_VT, (head << 5) + (nT << 4), 0);
                v16bf bf1 = load_frag_rowK(vt, LD_VT, (head << 5) + (nT << 4), 32);
                v8f acc = {};
                acc = wmma_bf16(pa[0], bf0, acc);
                acc = wmma_bf16(pa[1], bf1, acc);
                const int col = (head << 5) + (nT << 4) + (lane & 15);
                const int mo = (lane >> 4) << 3;
#pragma unroll
                for (int r2 = 0; r2 < 8; ++r2)
                    yb[((mT << 4) + mo + r2) * LD_Y + col] = f2bf(acc[r2]);
            }
        }
        __syncthreads();   // protect sc/pb before next pass
    }

    // ---- Phase 6: output projection + inverse roll + residual -> x1 ----
    {
        const int mT = wv & 3;
        v16bf afr[6];
#pragma unroll
        for (int kc = 0; kc < 6; ++kc)
            afr[kc] = load_frag_rowK(yb, LD_Y, mT << 4, kc << 5);
        for (int nT = (wv >> 2); nT < 12; nT += 2) {
            v16bf bfr[6];
#pragma unroll
            for (int kc = 0; kc < 6; ++kc)
                bfr[kc] = load_frag_rowK(wlin, C_DIM, nT << 4, kc << 5);
            v8f acc = {};
#pragma unroll
            for (int kc = 0; kc < 6; ++kc) acc = wmma_bf16(afr[kc], bfr[kc], acc);
            const int col = (nT << 4) + (lane & 15);
            const float bias = lin_b[col];
            const int mo = (lane >> 4) << 3;
#pragma unroll
            for (int r2 = 0; r2 < 8; ++r2) {
                const int p  = (mT << 4) + mo + r2;
                const int i1 = p >> 3, j1 = p & 7;
                const int hd2 = ((wh << 3) + i1 + SHIFT) & 63;
                const int wd2 = ((wc << 3) + j1 + SHIFT) & 63;
                const long gi = ((long)((bat << 6) + hd2) * 64 + wd2) * C_DIM + col;
                x1[gi] = x[gi] + acc[r2] + bias;
            }
        }
    }
}

// ---------------------------------------------------------------------------
// Kernel 2: LN2 + MLP (192 -> 768 -> 192, exact GELU) + residual -> out
// Padded LDS: zb u16[64][200] @0 (25600) | hb u16[64][776] @25600 (99328)
// total 124928 B
// ---------------------------------------------------------------------------
__global__ __launch_bounds__(256, 1)
void swin_mlp(const float* __restrict__ x1,
              const float* __restrict__ ln2_g, const float* __restrict__ ln2_b,
              const unsigned short* __restrict__ w1, const float* __restrict__ b1,
              const unsigned short* __restrict__ w2, const float* __restrict__ b2,
              float* __restrict__ out) {
    extern __shared__ __align__(16) char smem2[];
    unsigned short* zb = (unsigned short*)smem2;
    unsigned short* hb = (unsigned short*)(smem2 + 25600);

    const int tid = threadIdx.x;
    const int lane = tid & 31;
    const int wv = __builtin_amdgcn_readfirstlane(tid >> 5);
    const long t0 = (long)blockIdx.x * 64;

    // ---- LN2 (4 lanes per token) ----
    {
        const int tok = tid >> 2, part = tid & 3;
        const float4* s4 = (const float4*)(x1 + (t0 + tok) * C_DIM + part * 48);
        float4 vals[12];
        float s = 0.f, s2 = 0.f;
#pragma unroll
        for (int j = 0; j < 12; ++j) {
            float4 v = s4[j]; vals[j] = v;
            s  += v.x + v.y + v.z + v.w;
            s2 += v.x * v.x + v.y * v.y + v.z * v.z + v.w * v.w;
        }
        s  += __shfl_xor(s, 1);  s  += __shfl_xor(s, 2);
        s2 += __shfl_xor(s2, 1); s2 += __shfl_xor(s2, 2);
        const float m = s * (1.f / C_DIM);
        const float r = rsqrtf(s2 * (1.f / C_DIM) - m * m + 1e-5f);
        unsigned* dst = (unsigned*)(zb + tok * LD_Y + part * 48);
        const float4* g4 = (const float4*)(ln2_g + part * 48);
        const float4* b4 = (const float4*)(ln2_b + part * 48);
#pragma unroll
        for (int j = 0; j < 12; ++j) {
            float4 v = vals[j], g = g4[j], bb = b4[j];
            dst[2 * j]     = pack2((v.x - m) * r * g.x + bb.x, (v.y - m) * r * g.y + bb.y);
            dst[2 * j + 1] = pack2((v.z - m) * r * g.z + bb.z, (v.w - m) * r * g.w + bb.w);
        }
    }
    __syncthreads();

    // ---- GEMM1 (64x192)@(192x768) + exact GELU -> hb ----
    {
        const int mT = wv & 3;
        v16bf afr[6];
#pragma unroll
        for (int kc = 0; kc < 6; ++kc)
            afr[kc] = load_frag_rowK(zb, LD_Y, mT << 4, kc << 5);
        for (int nT = (wv >> 2); nT < 48; nT += 2) {
            v16bf bfr[6];
#pragma unroll
            for (int kc = 0; kc < 6; ++kc)
                bfr[kc] = load_frag_rowK(w1, C_DIM, nT << 4, kc << 5);
            v8f acc = {};
#pragma unroll
            for (int kc = 0; kc < 6; ++kc) acc = wmma_bf16(afr[kc], bfr[kc], acc);
            const int col = (nT << 4) + (lane & 15);
            const float bias = b1[col];
            const int mo = (lane >> 4) << 3;
#pragma unroll
            for (int r2 = 0; r2 < 8; ++r2) {
                float v = acc[r2] + bias;
                v = 0.5f * v * (1.f + erff(v * 0.70710678118654752f));
                hb[((mT << 4) + mo + r2) * LD_H + col] = f2bf(v);
            }
        }
    }
    __syncthreads();

    // ---- GEMM2 (64x768)@(768x192) + residual -> out ----
    {
        const int mT = wv & 3;
        for (int nT = (wv >> 2); nT < 12; nT += 2) {
            v8f acc = {};
#pragma unroll
            for (int kg = 0; kg < 4; ++kg) {
                v16bf af[6], bf6[6];
#pragma unroll
                for (int k6 = 0; k6 < 6; ++k6) {
                    const int kc = kg * 6 + k6;
                    af[k6]  = load_frag_rowK(hb, LD_H, mT << 4, kc << 5);
                    bf6[k6] = load_frag_rowK(w2, 768, nT << 4, kc << 5);
                }
#pragma unroll
                for (int k6 = 0; k6 < 6; ++k6) acc = wmma_bf16(af[k6], bf6[k6], acc);
            }
            const int col = (nT << 4) + (lane & 15);
            const float bias = b2[col];
            const int mo = (lane >> 4) << 3;
#pragma unroll
            for (int r2 = 0; r2 < 8; ++r2) {
                const long gi = (t0 + (mT << 4) + mo + r2) * C_DIM + col;
                out[gi] = x1[gi] + acc[r2] + bias;
            }
        }
    }
}

// ---------------------------------------------------------------------------
extern "C" void kernel_launch(void* const* d_in, const int* in_sizes, int n_in,
                              void* d_out, int out_size, void* d_ws, size_t ws_size,
                              hipStream_t stream) {
    (void)in_sizes; (void)n_in; (void)out_size; (void)ws_size;

    const float* x      = (const float*)d_in[0];
    const float* ln1_g  = (const float*)d_in[1];
    const float* ln1_b  = (const float*)d_in[2];
    const float* qkv_w  = (const float*)d_in[3];
    const float* qkv_b  = (const float*)d_in[4];
    const float* rpp    = (const float*)d_in[5];
    const float* lin_w  = (const float*)d_in[6];
    const float* lin_b  = (const float*)d_in[7];
    const float* ln2_g  = (const float*)d_in[8];
    const float* ln2_b  = (const float*)d_in[9];
    const float* mlp_w1 = (const float*)d_in[10];
    const float* mlp_b1 = (const float*)d_in[11];
    const float* mlp_w2 = (const float*)d_in[12];
    const float* mlp_b2 = (const float*)d_in[13];

    unsigned short* wsu  = (unsigned short*)d_ws;
    unsigned short* wqkv = wsu;
    unsigned short* wlin = wqkv + N_QKV;
    unsigned short* w1b  = wlin + N_LIN;
    unsigned short* w2b  = w1b + N_W1;
    float* x1 = (float*)((char*)d_ws + 2ull * (N_QKV + N_LIN + N_W1 + N_W2));

    const int nW = N_QKV + N_LIN + N_W1 + N_W2;
    prep_weights<<<(nW + 255) / 256, 256, 0, stream>>>(qkv_w, lin_w, mlp_w1, mlp_w2, wqkv);

    const int attnLds = 156672;
    const int mlpLds  = 124928;
    hipFuncSetAttribute((const void*)swin_attn,
                        hipFuncAttributeMaxDynamicSharedMemorySize, attnLds);
    hipFuncSetAttribute((const void*)swin_mlp,
                        hipFuncAttributeMaxDynamicSharedMemorySize, mlpLds);

    swin_attn<<<2048, 256, attnLds, stream>>>(x, ln1_g, ln1_b, wqkv, qkv_b, rpp,
                                              wlin, lin_b, x1);
    swin_mlp<<<2048, 256, mlpLds, stream>>>(x1, ln2_g, ln2_b, w1b, mlp_b1,
                                            w2b, mlp_b2, (float*)d_out);
}
